// ExpertGroup_1520418423057
// MI455X (gfx1250) — matrix operand order repaired and checked by reading, not compiled, not run on hardware
//
#include <hip/hip_runtime.h>
#include <hip/hip_bf16.h>

#define DM   1024
#define DH   2048
#define NE   16
#define NTOK 8192
#define TM   32            // tokens per tile
#define HPAD (DH + 8)      // LDS row pad (keeps 16B alignment, breaks bank stride)
#define NGRP 4             // N-tiles accumulated per wave group

#if __has_builtin(__builtin_amdgcn_global_load_async_to_lds_b128)
#define HAS_ASYNC 1
#else
#define HAS_ASYNC 0
#endif

typedef __attribute__((ext_vector_type(16))) __bf16       v16bf;
typedef __attribute__((ext_vector_type(8)))  float        v8f;
typedef __attribute__((ext_vector_type(4)))  unsigned int u32x4;
typedef __attribute__((ext_vector_type(4)))  int          v4i;

__device__ __forceinline__ unsigned short f2bf(float f) {
  unsigned int u = __float_as_uint(f);
  u += 0x7FFFu + ((u >> 16) & 1u);          // round-to-nearest-even
  return (unsigned short)(u >> 16);
}

// A fragment (16x32 bf16, M x K), row-major source with row stride ld (elements).
// ISA 7.12.2: lane L(<16): row L, K = {0..7, 16..23}; lane 16+L: row L, K = {8..15, 24..31}
__device__ __forceinline__ v16bf load_a_frag(const unsigned short* base, int ld, int lane) {
  const int lrow = lane & 15;
  const int kb   = (lane >> 4) << 3;        // 0 or 8
  const unsigned short* p = base + lrow * ld + kb;
  union { u32x4 q[2]; v16bf v; } u;
  u.q[0] = *(const u32x4*)(p);
  u.q[1] = *(const u32x4*)(p + 16);
  return u.v;
}

// B fragment (32x16 bf16, K x N) from pre-transposed [N][K] source, row stride ld.
// lane n(<16): col n, K = {0..7, 16..23} ; lane 16+n: col n, K = {8..15, 24..31}
__device__ __forceinline__ v16bf load_b_frag(const unsigned short* base, int ld, int lane) {
  const int lrow = lane & 15;
  const int kh   = (lane >> 4) << 3;        // 0 or 8
  const unsigned short* p = base + (long)lrow * (long)ld + kh;
  union { u32x4 q[2]; v16bf v; } u;
  u.q[0] = *(const u32x4*)(p);
  u.q[1] = *(const u32x4*)(p + 16);
  return u.v;
}

__device__ __forceinline__ v8f wmma_bf16(v16bf a, v16bf b, v8f c) {
  return __builtin_amdgcn_wmma_f32_16x16x32_bf16(false, a, false, b, (short)0, c, false, false);
}

// ---------------------------------------------------------------- zero / init
__global__ void zero_kernel(float* __restrict__ out, size_t n, int* __restrict__ counts) {
  size_t i = (size_t)blockIdx.x * blockDim.x + threadIdx.x;
  size_t stride = (size_t)gridDim.x * blockDim.x;
  for (; i < n; i += stride) out[i] = 0.0f;
  if (blockIdx.x == 0 && threadIdx.x < NE) counts[threadIdx.x] = 0;
}

// ---------------------------------------------------------------- cast fp32 -> bf16
__global__ void cast_kernel(const float* __restrict__ in, unsigned short* __restrict__ out, size_t n) {
  size_t i = (size_t)blockIdx.x * blockDim.x + threadIdx.x;
  size_t stride = (size_t)gridDim.x * blockDim.x;
  for (; i < n; i += stride) out[i] = f2bf(in[i]);
}

// ---------------------------------------------------------------- transpose+cast: [E][K][N] f32 -> [E][N][K] bf16
__global__ __launch_bounds__(256) void transpose_cast_kernel(
    const float* __restrict__ in, unsigned short* __restrict__ out, int K, int N) {
  __shared__ float tile[32][33];
  const int ntN = N >> 5, ntK = K >> 5;
  int bid = blockIdx.x;
  int e  = bid / (ntK * ntN);
  int r  = bid % (ntK * ntN);
  int kt = r / ntN, nt = r % ntN;
  int c  = threadIdx.x & 31, rr = threadIdx.x >> 5;     // 8 rows of 32
  const float* src = in + ((size_t)e * K + kt * 32) * N + nt * 32;
#pragma unroll
  for (int i = 0; i < 4; i++) tile[rr + 8 * i][c] = src[(size_t)(rr + 8 * i) * N + c];
  __syncthreads();
  unsigned short* dst = out + ((size_t)e * N + nt * 32) * K + kt * 32;
#pragma unroll
  for (int i = 0; i < 4; i++) dst[(size_t)(rr + 8 * i) * K + c] = f2bf(tile[c][rr + 8 * i]);
}

// ---------------------------------------------------------------- router: top-2 + softmax + bucket append
__global__ __launch_bounds__(128) void router_kernel(
    const float* __restrict__ x, const float* __restrict__ Wg,
    int* __restrict__ counts, int* __restrict__ btok, float* __restrict__ bw) {
  __shared__ float sx[DM];
  __shared__ float part[128];
  __shared__ float slog[NE];
  const int t = threadIdx.x;
  const int tok = blockIdx.x;
  const float* xr = x + (size_t)tok * DM;
  for (int i = t; i < DM; i += 128) sx[i] = xr[i];
  __syncthreads();
  const int e = t & 15, sl = t >> 4;                    // 16 experts x 8 slices of 128
  float acc = 0.0f;
  const float* wg = Wg + e * DM + sl * 128;
  const float* xs = sx + sl * 128;
#pragma unroll 4
  for (int i = 0; i < 128; i++) acc = fmaf(xs[i], wg[i], acc);
  part[t] = acc;
  __syncthreads();
  if (t < NE) {
    float s = 0.0f;
#pragma unroll
    for (int k = 0; k < 8; k++) s += part[t + 16 * k];
    slog[t] = s;
  }
  __syncthreads();
  if (t == 0) {
    float v0 = slog[0]; int i0 = 0;
    for (int i = 1; i < NE; i++) if (slog[i] > v0) { v0 = slog[i]; i0 = i; }   // first max (jax tie-break)
    float v1 = -3.4e38f; int i1 = 0;
    for (int i = 0; i < NE; i++) if (i != i0 && slog[i] > v1) { v1 = slog[i]; i1 = i; }
    float e1 = __expf(v1 - v0);
    float inv = 1.0f / (1.0f + e1);
    int p0 = atomicAdd(&counts[i0], 1);
    btok[i0 * NTOK + p0] = tok; bw[i0 * NTOK + p0] = inv;
    int p1 = atomicAdd(&counts[i1], 1);
    btok[i1 * NTOK + p1] = tok; bw[i1 * NTOK + p1] = e1 * inv;
  }
}

// ---------------------------------------------------------------- fused expert MLP (WMMA bf16)
__global__ __launch_bounds__(256) void moe_mlp_kernel(
    const unsigned short* __restrict__ xb,    // [NTOK][DM] bf16
    const unsigned short* __restrict__ W1t,   // [NE][DH][DM] bf16 (transposed)
    const float* __restrict__ b1,             // [NE][DH]
    const unsigned short* __restrict__ W2t,   // [NE][DM][DH] bf16 (transposed)
    const float* __restrict__ b2,             // [NE][DM]
    const int* __restrict__ counts,
    const int* __restrict__ btok,
    const float* __restrict__ bw,
    float* __restrict__ out) {                // [NTOK][DM]
  __shared__ unsigned short sX[TM * DM];      // 64 KB
  __shared__ unsigned short sH[TM * HPAD];    // ~128.5 KB  (CDNA5 big LDS)
  __shared__ int   stok[TM];
  __shared__ float sw[TM];

  const int e    = blockIdx.x >> 8;           // 256 tiles per expert
  const int tile = blockIdx.x & 255;
  const int cnt  = counts[e];
  const int t0   = tile * TM;
  if (t0 >= cnt) return;

  const int tid = threadIdx.x;
  if (tid < TM) {
    int idx = t0 + tid;
    stok[tid] = (idx < cnt) ? btok[e * NTOK + idx] : -1;
    sw[tid]   = (idx < cnt) ? bw[e * NTOK + idx] : 0.0f;
  }
  __syncthreads();

  // gather X tile (32 rows x 1024 bf16) as 16B chunks (async direct-to-LDS if available)
  for (int j = tid; j < TM * DM / 8; j += 256) {
    int row = j >> 7, col = j & 127;          // 128 x 16B per row
    int tk = stok[row]; if (tk < 0) tk = 0;
    v4i* src = (v4i*)(xb + (size_t)tk * DM) + col;
#if HAS_ASYNC
    __builtin_amdgcn_global_load_async_to_lds_b128(src, ((v4i*)sX) + j, 0, 0);
#else
    ((u32x4*)sX)[j] = *(const u32x4*)src;
#endif
  }
#if HAS_ASYNC
  asm volatile("s_wait_asynccnt 0x0" ::: "memory");
#endif
  __syncthreads();

  const int wave = tid >> 5;
  const int lane = tid & 31;
  const int lrow = lane & 15;
  const int lhi  = lane >> 4;

  // ---------------- GEMM1: H = relu(X @ W1[e] + b1[e])  -> sH (bf16)
  const unsigned short* W1e = W1t + (size_t)e * DH * DM;
  for (int ng = wave; ng < (DH / 16) / NGRP; ng += 8) {   // 32 groups of 4 N-tiles
    const int n0 = ng * (16 * NGRP);
    v8f acc0[NGRP], acc1[NGRP];
#pragma unroll
    for (int t = 0; t < NGRP; t++) { acc0[t] = (v8f){}; acc1[t] = (v8f){}; }
#pragma unroll 2
    for (int k0 = 0; k0 < DM; k0 += 32) {
      v16bf a0 = load_a_frag(sX + k0, DM, lane);
      v16bf a1 = load_a_frag(sX + 16 * DM + k0, DM, lane);
#pragma unroll
      for (int t = 0; t < NGRP; t++) {
        v16bf bb = load_b_frag(W1e + (size_t)(n0 + 16 * t) * DM + k0, DM, lane);
        acc0[t] = wmma_bf16(a0, bb, acc0[t]);
        acc1[t] = wmma_bf16(a1, bb, acc1[t]);
      }
    }
#pragma unroll
    for (int t = 0; t < NGRP; t++) {
      const int nc = n0 + 16 * t + lrow;
      const float bias = b1[e * DH + nc];
#pragma unroll
      for (int j = 0; j < 8; j++) {
        const int m = j + lhi * 8;
        sH[m * HPAD + nc]        = f2bf(fmaxf(acc0[t][j] + bias, 0.0f));
        sH[(m + 16) * HPAD + nc] = f2bf(fmaxf(acc1[t][j] + bias, 0.0f));
      }
    }
  }
  __syncthreads();

  // ---------------- GEMM2: Y = H @ W2[e] + b2[e], scaled by token weight, accumulated
  const unsigned short* W2e = W2t + (size_t)e * DM * DH;
  for (int ng = wave; ng < (DM / 16) / NGRP; ng += 8) {   // 16 groups of 4 N-tiles
    const int n0 = ng * (16 * NGRP);
    v8f acc0[NGRP], acc1[NGRP];
#pragma unroll
    for (int t = 0; t < NGRP; t++) { acc0[t] = (v8f){}; acc1[t] = (v8f){}; }
#pragma unroll 2
    for (int k0 = 0; k0 < DH; k0 += 32) {
      v16bf a0 = load_a_frag(sH + k0, HPAD, lane);
      v16bf a1 = load_a_frag(sH + 16 * HPAD + k0, HPAD, lane);
#pragma unroll
      for (int t = 0; t < NGRP; t++) {
        v16bf bb = load_b_frag(W2e + (size_t)(n0 + 16 * t) * DH + k0, DH, lane);
        acc0[t] = wmma_bf16(a0, bb, acc0[t]);
        acc1[t] = wmma_bf16(a1, bb, acc1[t]);
      }
    }
#pragma unroll
    for (int t = 0; t < NGRP; t++) {
      const int nc = n0 + 16 * t + lrow;
      const float bias = b2[e * DM + nc];
#pragma unroll
      for (int j = 0; j < 8; j++) {
        const int m = j + lhi * 8;
        int tk0 = stok[m];
        if (tk0 >= 0) atomicAdd(out + (size_t)tk0 * DM + nc, (acc0[t][j] + bias) * sw[m]);
        int tk1 = stok[m + 16];
        if (tk1 >= 0) atomicAdd(out + (size_t)tk1 * DM + nc, (acc1[t][j] + bias) * sw[m + 16]);
      }
    }
  }
}

// ---------------------------------------------------------------- launch
extern "C" void kernel_launch(void* const* d_in, const int* in_sizes, int n_in,
                              void* d_out, int out_size, void* d_ws, size_t ws_size,
                              hipStream_t stream) {
  const float* x  = (const float*)d_in[0];
  const float* Wg = (const float*)d_in[1];
  const float* W1 = (const float*)d_in[2];
  const float* b1 = (const float*)d_in[3];
  const float* W2 = (const float*)d_in[4];
  const float* b2 = (const float*)d_in[5];
  float* out = (float*)d_out;

  char* w = (char*)d_ws;
  unsigned short* xb  = (unsigned short*)w; w += (size_t)NTOK * DM * 2;        // 16 MB
  unsigned short* W1t = (unsigned short*)w; w += (size_t)NE * DH * DM * 2;     // 64 MB
  unsigned short* W2t = (unsigned short*)w; w += (size_t)NE * DM * DH * 2;     // 64 MB
  int*   counts = (int*)w;  w += 256;
  int*   btok   = (int*)w;  w += (size_t)NE * NTOK * 4;
  float* bwp    = (float*)w;

  zero_kernel<<<4096, 256, 0, stream>>>(out, (size_t)NTOK * DM, counts);
  cast_kernel<<<4096, 256, 0, stream>>>(x, xb, (size_t)NTOK * DM);
  transpose_cast_kernel<<<NE * (DM / 32) * (DH / 32), 256, 0, stream>>>(W1, W1t, DM, DH);
  transpose_cast_kernel<<<NE * (DH / 32) * (DM / 32), 256, 0, stream>>>(W2, W2t, DH, DM);
  router_kernel<<<NTOK, 128, 0, stream>>>(x, Wg, counts, btok, bwp);
  moe_mlp_kernel<<<NE * 256, 256, 0, stream>>>(xb, W1t, b1, W2t, b2, counts, btok, bwp, out);
}
